// MaskPredictor_41351945125991
// MI455X (gfx1250) — matrix operand order repaired and checked by reading, not compiled, not run on hardware
//
#include <hip/hip_runtime.h>
#include <hip/hip_bf16.h>
#include <math.h>

typedef __attribute__((ext_vector_type(16))) _Float16 v16h;
typedef __attribute__((ext_vector_type(8)))  _Float16 v8h;
typedef __attribute__((ext_vector_type(4)))  _Float16 v4h;
typedef __attribute__((ext_vector_type(8)))  float    v8f;

#define WAVES_PER_BLOCK 8
#define THREADS (WAVES_PER_BLOCK * 32)
#define LDS_STRIDE 264   // 256 halves + 8 pad (spreads LDS banks)

// Pass 1: stream-convert node features f32 -> f16 into workspace.
__global__ __launch_bounds__(256)
void convert_z_f16(const float* __restrict__ z, _Float16* __restrict__ zh, int n4)
{
    int i = blockIdx.x * blockDim.x + threadIdx.x;
    if (i < n4) {
        float4 u = ((const float4*)z)[i];
        v4h h;
        h[0] = (_Float16)u.x; h[1] = (_Float16)u.y;
        h[2] = (_Float16)u.z; h[3] = (_Float16)u.w;
        ((v4h*)zh)[i] = h;
    }
}

// Pass 2: edge MLP, M=16 edges/tile, K=256 (8 WMMA steps), N=64 (4 c-tiles).
// B (W1^T) staged to LDS, register-hoisted by the compiler. The whole A tile
// (16x global_load_b128) is kept live so loads issue as one deep clause with
// staggered waits, and the NEXT tile's A loads are issued right after the
// current tile's WMMAs -> cross-tile software pipeline with 16 loads in flight.
template <bool HALFZ>
__global__ __launch_bounds__(THREADS)
void mask_predictor_edge_mlp(const float* __restrict__ z,        // [N,128] f32
                             const _Float16* __restrict__ zh,    // [N,128] f16 (ws)
                             const int* __restrict__ ei32,       // [2,E] int64 as int32 pairs
                             const float* __restrict__ W1,       // [256, 64]
                             const float* __restrict__ b1,       // [64]
                             const float* __restrict__ W2,       // [64]
                             const float* __restrict__ b2,       // [1]
                             float* __restrict__ out,            // [E]
                             int E, int ntiles)
{
    __shared__ _Float16 w1t[64 * LDS_STRIDE];   // W1^T as f16: w1t[n*264 + k]

    for (int i = threadIdx.x; i < 256 * 64; i += THREADS) {
        int k = i >> 6, n = i & 63;
        w1t[n * LDS_STRIDE + k] = (_Float16)W1[i];
    }
    __syncthreads();

    const int lane = threadIdx.x & 31;
    const int wave = threadIdx.x >> 5;
    const int row  = lane & 15;   // edge-in-tile (A rows) and N-in-tile (B/C cols)
    const int hi   = lane >> 4;   // lane-half selector

    // Per-lane layer-2 constants: this lane owns hidden column nt*16+row
    float b1v[4], w2v[4];
    #pragma unroll
    for (int nt = 0; nt < 4; ++nt) {
        b1v[nt] = b1[nt * 16 + row];
        w2v[nt] = W2[nt * 16 + row];
    }
    const float b2v = b2[0];

    const int stride = gridDim.x * WAVES_PER_BLOCK;
    int tile = blockIdx.x * WAVES_PER_BLOCK + wave;
    if (tile >= ntiles) return;   // wave-uniform

    // A-fragment loader (ISA 16-bit A 16x32 layout): runs kb+hi*8+[0,8) and +16.
    auto loadA = [&](unsigned sO, unsigned dO, int ks) -> v16h {
        const unsigned nOff = (ks < 4) ? sO : dO;
        const int kb = (ks & 3) * 32 + hi * 8;
        if (HALFZ) {
            const _Float16* zp = zh + nOff + kb;
            v8h lo = *(const v8h*)zp;          // global_load_b128
            v8h h8 = *(const v8h*)(zp + 16);   // global_load_b128
            return __builtin_shufflevector(lo, h8,
                   0,1,2,3,4,5,6,7,8,9,10,11,12,13,14,15);
        } else {
            const float* zp = z + nOff + kb;
            float4 u0 = ((const float4*)zp)[0];
            float4 u1 = ((const float4*)zp)[1];
            float4 u2 = ((const float4*)(zp + 16))[0];
            float4 u3 = ((const float4*)(zp + 16))[1];
            float fa[16] = { u0.x,u0.y,u0.z,u0.w, u1.x,u1.y,u1.z,u1.w,
                             u2.x,u2.y,u2.z,u2.w, u3.x,u3.y,u3.z,u3.w };
            v16h a;
            #pragma unroll
            for (int i = 0; i < 16; ++i) a[i] = (_Float16)fa[i];
            return a;
        }
    };
    auto loadIdx = [&](int t, unsigned& sO, unsigned& dO) {
        int e = t * 16 + row; if (e >= E) e = E - 1;
        sO = (unsigned)ei32[2 * e] * 128u;                  // low word of int64
        dO = (unsigned)ei32[2 * ((size_t)E + e)] * 128u;
    };

    // Prologue: indices + full A tile for the first tile.
    unsigned sCur, dCur;
    loadIdx(tile, sCur, dCur);
    v16h atile[8];
    #pragma unroll
    for (int ks = 0; ks < 8; ++ks) atile[ks] = loadA(sCur, dCur, ks);

    for (; tile < ntiles; tile += stride) {
        // Issue NEXT tile's index loads now; consumed ~32 WMMAs later.
        unsigned sNxt, dNxt;
        {
            int tn = (tile + stride < ntiles) ? (tile + stride) : tile;
            loadIdx(tn, sNxt, dNxt);
        }

        v8f c0 = {}, c1 = {}, c2 = {}, c3 = {};
        #pragma unroll
        for (int ks = 0; ks < 8; ++ks) {
            const v16h a = atile[ks];

            // B fragments: lane = col N, K runs hi*8+[0,8) and +16 (register-hoisted)
            const int kwb = ks * 32 + hi * 8;
            v16h bb[4];
            #pragma unroll
            for (int nt = 0; nt < 4; ++nt) {
                const _Float16* wp = &w1t[(nt * 16 + row) * LDS_STRIDE + kwb];
                v8h lo  = *(const v8h*)wp;
                v8h hi8 = *(const v8h*)(wp + 16);
                bb[nt] = __builtin_shufflevector(lo, hi8,
                         0,1,2,3,4,5,6,7,8,9,10,11,12,13,14,15);
            }

            c0 = __builtin_amdgcn_wmma_f32_16x16x32_f16(false, a, false, bb[0], (short)0, c0, false, false);
            c1 = __builtin_amdgcn_wmma_f32_16x16x32_f16(false, a, false, bb[1], (short)0, c1, false, false);
            c2 = __builtin_amdgcn_wmma_f32_16x16x32_f16(false, a, false, bb[2], (short)0, c2, false, false);
            c3 = __builtin_amdgcn_wmma_f32_16x16x32_f16(false, a, false, bb[3], (short)0, c3, false, false);
        }

        // Issue the NEXT tile's 16 A gathers; they fly during layer 2 + stores
        // and are consumed next iteration with staggered s_wait_loadcnt.
        #pragma unroll
        for (int ks = 0; ks < 8; ++ks) atile[ks] = loadA(sNxt, dNxt, ks);

        // ----- Layer 2: bias + ReLU + dot(W2) in registers -----
        // C layout: lane holds N = row; VGPR r holds M = r + 8*hi.
        float acc[8];
        #pragma unroll
        for (int r = 0; r < 8; ++r) {
            float s;
            s  = fmaxf(c0[r] + b1v[0], 0.f) * w2v[0];
            s += fmaxf(c1[r] + b1v[1], 0.f) * w2v[1];
            s += fmaxf(c2[r] + b1v[2], 0.f) * w2v[2];
            s += fmaxf(c3[r] + b1v[3], 0.f) * w2v[3];
            acc[r] = s;
        }
        // Butterfly-reduce over the 16 N-lanes (masks <=8 stay within each half)
        #pragma unroll
        for (int off = 1; off < 16; off <<= 1) {
            #pragma unroll
            for (int r = 0; r < 8; ++r) acc[r] += __shfl_xor(acc[r], off, 32);
        }

        if (row == 0) {   // lanes 0 (M=0..7) and 16 (M=8..15) write results
            const int base = tile * 16 + hi * 8;
            #pragma unroll
            for (int r = 0; r < 8; ++r) {
                int m = base + r;
                if (m < E) out[m] = 1.f / (1.f + __expf(-(acc[r] + b2v)));
            }
        }
    }
}

extern "C" void kernel_launch(void* const* d_in, const int* in_sizes, int n_in,
                              void* d_out, int out_size, void* d_ws, size_t ws_size,
                              hipStream_t stream) {
    const float* z    = (const float*)d_in[0];
    const int*   ei32 = (const int*)d_in[1];   // int64 [2,E] viewed as int32 pairs
    const float* W1   = (const float*)d_in[2];
    const float* b1   = (const float*)d_in[3];
    const float* W2   = (const float*)d_in[4];
    const float* b2   = (const float*)d_in[5];
    float*       out  = (float*)d_out;

    const int E      = in_sizes[1] / 2;
    const int zElems = in_sizes[0];            // N_NODES * 128
    const int ntiles = (E + 15) / 16;
    int blocks = (ntiles + WAVES_PER_BLOCK - 1) / WAVES_PER_BLOCK;
    if (blocks > 1024) blocks = 1024;          // persistent waves; amortize staging

    const bool halfz = ws_size >= (size_t)zElems * sizeof(_Float16);
    if (halfz) {
        _Float16* zh = (_Float16*)d_ws;
        const int n4 = zElems / 4;
        convert_z_f16<<<(n4 + 255) / 256, 256, 0, stream>>>(z, zh, n4);
        mask_predictor_edge_mlp<true><<<blocks, THREADS, 0, stream>>>(
            z, zh, ei32, W1, b1, W2, b2, out, E, ntiles);
    } else {
        mask_predictor_edge_mlp<false><<<blocks, THREADS, 0, stream>>>(
            z, (const _Float16*)nullptr, ei32, W1, b1, W2, b2, out, E, ntiles);
    }
}
